// LSTM_14637248545159
// MI455X (gfx1250) — matrix operand order repaired and checked by reading, not compiled, not run on hardware
//
#include <hip/hip_runtime.h>
#include <hip/hip_bf16.h>

// ---------------------------------------------------------------------------
// CDNA5 / gfx1250 LSTM stack: persistent per-batch-tile workgroups, WMMA f16
// ---------------------------------------------------------------------------

typedef __attribute__((ext_vector_type(16))) _Float16 v16h;
typedef __attribute__((ext_vector_type(8)))  float    v8f;

union FragU {
    v16h h;
    uint4 u[2];
};

#define B_TOT 1024
#define T_TOT 256

// ---------------------------------------------------------------------------
// Pack [W_ih ; W_hh]^T into WMMA B-fragment order (f16), fold biases.
// B tile as packed here (32x16, contiguous k, matching A): lane l holds column N = l&15, K values
// ktile*32 + ((l&16)?16:0) + j  for j=0..15, contiguous per lane (32 bytes).
// ---------------------------------------------------------------------------
template<int IN_DIM, int H>
__global__ void pack_weights_kernel(const float* __restrict__ w_ih,
                                    const float* __restrict__ w_hh,
                                    const float* __restrict__ b_ih,
                                    const float* __restrict__ b_hh,
                                    _Float16* __restrict__ wcat,
                                    float* __restrict__ bsum)
{
    constexpr int N4H = 4 * H;
    constexpr int KP  = ((IN_DIM + H + 31) / 32) * 32;
    constexpr int KS  = KP / 32;
    constexpr int NT  = N4H / 16;
    const int total = KS * NT * 512;        // halves

    for (int idx = blockIdx.x * blockDim.x + threadIdx.x; idx < total;
         idx += gridDim.x * blockDim.x) {
        int frag   = idx >> 9;              // which 32x16 fragment
        int within = idx & 511;
        int lane   = within >> 4;
        int j      = within & 15;
        int kt     = frag / NT;
        int nt     = frag - kt * NT;
        int k      = kt * 32 + ((lane & 16) ? 16 : 0) + j;
        int n      = nt * 16 + (lane & 15);
        float v = 0.0f;
        if (k < IN_DIM)            v = w_ih[n * IN_DIM + k];
        else if (k < IN_DIM + H)   v = w_hh[n * H + (k - IN_DIM)];
        wcat[idx] = (_Float16)v;
    }
    for (int i = blockIdx.x * blockDim.x + threadIdx.x; i < N4H;
         i += gridDim.x * blockDim.x)
        bsum[i] = b_ih[i] + b_hh[i];
}

// ---------------------------------------------------------------------------
// Persistent LSTM layer kernel. One WG owns BM=32 batch rows for all T steps.
// Per step: gates[BM][4H] = [x_t , h] @ Wcat  (WMMA f16 -> f32 accum),
// then gate nonlinearity updates c (LDS f32) and h (LDS f16 + global f16).
// ---------------------------------------------------------------------------
template<int IN_DIM, int H, bool XF32>
__global__ __launch_bounds__(256)
void lstm_layer_kernel(const void* __restrict__ xin_v,
                       const _Float16* __restrict__ wcat,
                       const float* __restrict__ bsum,
                       _Float16* __restrict__ hout)
{
    constexpr int BM  = 32;
    constexpr int N4H = 4 * H;
    constexpr int KP  = ((IN_DIM + H + 31) / 32) * 32;
    constexpr int KS  = KP / 32;
    constexpr int NT  = N4H / 16;
    constexpr int NTW = NT / 4;             // tiles per wave (8 waves, 2 m-tiles)

    extern __shared__ __align__(16) unsigned char smem[];
    _Float16* Abuf = (_Float16*)smem;                                   // BM*KP f16
    float*    Cst  = (float*)(smem + (size_t)BM * KP * 2);              // BM*H f32
    float*    Gat  = (float*)(smem + (size_t)BM * KP * 2
                                   + (size_t)BM * H  * 4);              // BM*N4H f32
    float*    Bs   = (float*)(smem + (size_t)BM * KP * 2
                                   + (size_t)BM * H  * 4
                                   + (size_t)BM * N4H * 4);             // N4H f32

    const int tid  = threadIdx.x;
    const int lane = tid & 31;
    const int wave = tid >> 5;
    const int b0   = blockIdx.x * BM;

    // zero A buffer (covers K padding + initial h=0), c=0, stage biases
    for (int i = tid; i < BM * KP; i += 256) Abuf[i] = (_Float16)0.0f;
    for (int i = tid; i < BM * H;  i += 256) Cst[i]  = 0.0f;
    for (int i = tid; i < N4H;     i += 256) Bs[i]   = bsum[i];
    __syncthreads();

    const int mt   = wave & 1;       // which 16-row M tile
    const int ng   = wave >> 1;      // which group of N tiles
    // A fragment as loaded here (16x32, contiguous k): lane l (0..15) row M=l, K = k0+{0..15};
    // lane l+16 same row, K = k0+{16..31}
    const int arow    = mt * 16 + (lane & 15);
    const int acolsel = (lane & 16) ? 16 : 0;

    const float*    xf = (const float*)xin_v;
    const _Float16* xh = (const _Float16*)xin_v;

    for (int t = 0; t < T_TOT; ++t) {
        // ---- phase 1: stage x_t tile into A buffer cols [0, IN_DIM) -------
        for (int i = tid; i < BM * IN_DIM; i += 256) {
            int m  = i / IN_DIM;
            int cc = i - m * IN_DIM;
            size_t gidx = ((size_t)(b0 + m) * T_TOT + t) * IN_DIM + cc;
            float v = XF32 ? xf[gidx] : (float)xh[gidx];
            Abuf[m * KP + cc] = (_Float16)v;
        }
        __syncthreads();

        // ---- phase 2: gates = [x_t , h] @ Wcat  (WMMA) --------------------
        v8f acc[NTW];
        const v8f vzero = {0.0f, 0.0f, 0.0f, 0.0f, 0.0f, 0.0f, 0.0f, 0.0f};
        #pragma unroll
        for (int i = 0; i < NTW; ++i) acc[i] = vzero;

        for (int k = 0; k < KS; ++k) {
            FragU a;
            const _Float16* ap = Abuf + arow * KP + k * 32 + acolsel;
            a.u[0] = *(const uint4*)(ap);
            a.u[1] = *(const uint4*)(ap + 8);

            const _Float16* bp0 =
                wcat + ((size_t)(k * NT + ng * NTW) * 32 + lane) * 16;
            if (k + 1 < KS)   // stream next K-slice of weights through caches
                __builtin_prefetch(
                    wcat + ((size_t)((k + 1) * NT + ng * NTW) * 32 + lane) * 16,
                    0, 0);

            #pragma unroll
            for (int nti = 0; nti < NTW; ++nti) {
                FragU b;
                const uint4* bp = (const uint4*)(bp0 + nti * 512);
                b.u[0] = bp[0];
                b.u[1] = bp[1];
                acc[nti] = __builtin_amdgcn_wmma_f32_16x16x32_f16(
                    false, a.h, false, b.h, (short)0, acc[nti], false, false);
            }
        }

        // spill accumulators to LDS gate buffer
        // C layout: VGPR r -> lanes 0..15: (M=r, N=lane); lanes 16..31: (M=r+8)
        const int crow0 = mt * 16 + ((lane & 16) ? 8 : 0);
        #pragma unroll
        for (int nti = 0; nti < NTW; ++nti) {
            int col = (ng * NTW + nti) * 16 + (lane & 15);
            #pragma unroll
            for (int r = 0; r < 8; ++r)
                Gat[(crow0 + r) * N4H + col] = acc[nti][r];
        }
        __syncthreads();

        // ---- phase 3: gate nonlinearity, state update ---------------------
        for (int i = tid; i < BM * H; i += 256) {
            int m = i / H;
            int j = i - m * H;
            float gi = Gat[m * N4H + j]         + Bs[j];
            float gf = Gat[m * N4H + H + j]     + Bs[H + j];
            float gg = Gat[m * N4H + 2 * H + j] + Bs[2 * H + j];
            float go = Gat[m * N4H + 3 * H + j] + Bs[3 * H + j];
            float si = 1.0f / (1.0f + __expf(-gi));
            float sf = 1.0f / (1.0f + __expf(-gf));
            float so = 1.0f / (1.0f + __expf(-go));
            float c  = sf * Cst[i] + si * tanhf(gg);
            float h  = so * tanhf(c);
            Cst[i] = c;
            Abuf[m * KP + IN_DIM + j] = (_Float16)h;   // feed next step's GEMM
            hout[((size_t)(b0 + m) * T_TOT + t) * H + j] = (_Float16)h;
        }
        __syncthreads();
    }
}

// ---------------------------------------------------------------------------
// FC head: h(t=255) -> fc1(32->8) -> fc2(8->1) -> sigmoid
// ---------------------------------------------------------------------------
__global__ void fc_head_kernel(const _Float16* __restrict__ h4,
                               const float* __restrict__ fc1w,
                               const float* __restrict__ fc1b,
                               const float* __restrict__ fc2w,
                               const float* __restrict__ fc2b,
                               float* __restrict__ out)
{
    int b = blockIdx.x * blockDim.x + threadIdx.x;
    if (b >= B_TOT) return;
    const _Float16* h = h4 + ((size_t)b * T_TOT + 255) * 32;
    float hl[32];
    #pragma unroll
    for (int k = 0; k < 32; ++k) hl[k] = (float)h[k];
    float acc2 = fc2b[0];
    #pragma unroll
    for (int j = 0; j < 8; ++j) {
        float a = fc1b[j];
        #pragma unroll
        for (int k = 0; k < 32; ++k) a += fc1w[j * 32 + k] * hl[k];
        acc2 += fc2w[j] * a;
    }
    out[b] = 1.0f / (1.0f + __expf(-acc2));
}

// ---------------------------------------------------------------------------
extern "C" void kernel_launch(void* const* d_in, const int* in_sizes, int n_in,
                              void* d_out, int out_size, void* d_ws, size_t ws_size,
                              hipStream_t stream)
{
    (void)in_sizes; (void)n_in; (void)out_size; (void)ws_size;

    const float* x     = (const float*)d_in[0];
    const float* w_ih1 = (const float*)d_in[1];
    const float* w_hh1 = (const float*)d_in[2];
    const float* b_ih1 = (const float*)d_in[3];
    const float* b_hh1 = (const float*)d_in[4];
    const float* w_ih2 = (const float*)d_in[5];
    const float* w_hh2 = (const float*)d_in[6];
    const float* b_ih2 = (const float*)d_in[7];
    const float* b_hh2 = (const float*)d_in[8];
    const float* w_ih3 = (const float*)d_in[9];
    const float* w_hh3 = (const float*)d_in[10];
    const float* b_ih3 = (const float*)d_in[11];
    const float* b_hh3 = (const float*)d_in[12];
    const float* w_ih4 = (const float*)d_in[13];
    const float* w_hh4 = (const float*)d_in[14];
    const float* b_ih4 = (const float*)d_in[15];
    const float* b_hh4 = (const float*)d_in[16];
    const float* fc1w  = (const float*)d_in[17];
    const float* fc1b  = (const float*)d_in[18];
    const float* fc2w  = (const float*)d_in[19];
    const float* fc2b  = (const float*)d_in[20];
    float* out = (float*)d_out;

    // ---- workspace carve-out (256B aligned) -------------------------------
    char* ws = (char*)d_ws;
    size_t o = 0;
    auto take = [&](size_t bytes) -> char* {
        char* p = ws + o;
        o = (o + bytes + 255) & ~(size_t)255;
        return p;
    };
    _Float16* W1 = (_Float16*)take((size_t)320 * 1024 * 2);   // KP1*4H1
    _Float16* W2 = (_Float16*)take((size_t)384 * 512  * 2);
    _Float16* W3 = (_Float16*)take((size_t)192 * 256  * 2);
    _Float16* W4 = (_Float16*)take((size_t)96  * 128  * 2);
    float* bs1 = (float*)take(1024 * 4);
    float* bs2 = (float*)take(512  * 4);
    float* bs3 = (float*)take(256  * 4);
    float* bs4 = (float*)take(128  * 4);
    _Float16* H1o = (_Float16*)take((size_t)B_TOT * T_TOT * 256 * 2);
    _Float16* H2o = (_Float16*)take((size_t)B_TOT * T_TOT * 128 * 2);
    _Float16* H3o = (_Float16*)take((size_t)B_TOT * T_TOT * 64  * 2);
    _Float16* H4o = (_Float16*)take((size_t)B_TOT * T_TOT * 32  * 2);

    // ---- weight packing ---------------------------------------------------
    pack_weights_kernel<34, 256><<<1024, 256, 0, stream>>>(w_ih1, w_hh1, b_ih1, b_hh1, W1, bs1);
    pack_weights_kernel<256,128><<<1024, 256, 0, stream>>>(w_ih2, w_hh2, b_ih2, b_hh2, W2, bs2);
    pack_weights_kernel<128, 64><<<512,  256, 0, stream>>>(w_ih3, w_hh3, b_ih3, b_hh3, W3, bs3);
    pack_weights_kernel<64,  32><<<128,  256, 0, stream>>>(w_ih4, w_hh4, b_ih4, b_hh4, W4, bs4);

    // ---- persistent LSTM layers ------------------------------------------
    // dynamic LDS: BM*KP*2 + BM*H*4 + BM*4H*4 + 4H*4
    constexpr size_t smem1 = 32 * 320 * 2 + 32 * 256 * 4 + 32 * 1024 * 4 + 1024 * 4; // 188416
    constexpr size_t smem2 = 32 * 384 * 2 + 32 * 128 * 4 + 32 * 512  * 4 + 512  * 4; // 108544
    constexpr size_t smem3 = 32 * 192 * 2 + 32 * 64  * 4 + 32 * 256  * 4 + 256  * 4; //  54272
    constexpr size_t smem4 = 32 * 96  * 2 + 32 * 32  * 4 + 32 * 128  * 4 + 128  * 4; //  27136

    lstm_layer_kernel<34,  256, true ><<<B_TOT / 32, 256, smem1, stream>>>(x,   W1, bs1, H1o);
    lstm_layer_kernel<256, 128, false><<<B_TOT / 32, 256, smem2, stream>>>(H1o, W2, bs2, H2o);
    lstm_layer_kernel<128, 64,  false><<<B_TOT / 32, 256, smem3, stream>>>(H2o, W3, bs3, H3o);
    lstm_layer_kernel<64,  32,  false><<<B_TOT / 32, 256, smem4, stream>>>(H3o, W4, bs4, H4o);

    // ---- FC head ----------------------------------------------------------
    fc_head_kernel<<<B_TOT / 256, 256, 0, stream>>>(H4o, fc1w, fc1b, fc2w, fc2b, out);
}
